// NNConvEncoder_8375186227331
// MI455X (gfx1250) — matrix operand order repaired and checked by reading, not compiled, not run on hardware
//
#include <hip/hip_runtime.h>
#include <hip/hip_bf16.h>

// ---------------------------------------------------------------------------
// NNConv (edge-conditioned conv) + GRU + Set2Set encoder for MI455X (gfx1250).
//
// msg_e = z_e @ W2v + x_e @ B2m with z_e = vec(outer(hmid_e, x_e)) formed on
// the fly in f16 registers; W2v is LDS-resident. Each workgroup owns
// (2 output 16-col tiles) x (half of K) = 260KB LDS slab, so every on-the-fly
// A operand feeds TWO v_wmma_f32_16x16x32_f16. A sched_barrier (SALU/VMEM-rd/
// DS-rd transparent) pins both half-A builds ahead of the 4-WMMA group so the
// builds live in distinct registers and the next iteration's v_pk_mul_f16
// stream fills the WMMA->VALU WAR co-exec window (no v_nops). B slab staged
// via global_load_async_to_lds_b128 + s_wait_asynccnt; K-half partial sums
// merge through the fused atomic scatter-aggregate.
// ---------------------------------------------------------------------------

typedef _Float16 v8h  __attribute__((ext_vector_type(8)));
typedef _Float16 v16h __attribute__((ext_vector_type(16)));
typedef float    v8f  __attribute__((ext_vector_type(8)));

#define HID   64
#define KH    128              // edge-MLP hidden
#define NCL   130              // chunks per K-half slab (128 real + 2 bias/pad)
#define CHKH  512              // f16 per chunk: 16 cols * 32 K

// sched_barrier transparency mask: SALU | VMEM-read | DS-read may cross;
// VALU must not sink past it, WMMA must not hoist above it.
#define SB_MASK 0x124

__device__ __forceinline__ float sigm(float x) { return 1.0f / (1.0f + __expf(-x)); }

__device__ __forceinline__ v8h scale8(v8h x, _Float16 s) {
    v8h r;
#pragma unroll
    for (int j = 0; j < 8; ++j) r[j] = x[j] * s;   // lowers to v_pk_mul_f16 (op_sel)
    return r;
}

__device__ __forceinline__ v16h ldB16(const _Float16* bp) {
    v8h b0 = *(const v8h*)bp;                      // ds_load_b128
    v8h b1 = *(const v8h*)(bp + 16);               // ds_load_b128
    return __builtin_shufflevector(b0, b1, 0,1,2,3,4,5,6,7,8,9,10,11,12,13,14,15);
}

// ---------------------------------------------------------------------------
// Small utility kernels
// ---------------------------------------------------------------------------
__global__ void k_zero_f32(float* p, long n) {
    long i = (long)blockIdx.x * blockDim.x + threadIdx.x;
    for (; i < n; i += (long)gridDim.x * blockDim.x) p[i] = 0.0f;
}

// h = relu(nfeat @ lin0_w + lin0_b)   [N,16]x[16,64]
__global__ void k_lin0(const float* __restrict__ nfeat, const float* __restrict__ w,
                       const float* __restrict__ b, float* __restrict__ h, int N) {
    long i = (long)blockIdx.x * blockDim.x + threadIdx.x;
    long tot = (long)N * HID;
    for (; i < tot; i += (long)gridDim.x * blockDim.x) {
        int node = (int)(i >> 6), col = (int)(i & 63);
        float s = b[col];
#pragma unroll
        for (int j = 0; j < 16; ++j) s += nfeat[node * 16 + j] * w[j * HID + col];
        h[i] = fmaxf(s, 0.0f);
    }
}

// hmid16 = f16(relu(efeat @ e1_w + e1_b))   [E,5]x[5,128], stored f16 (WMMA A feed)
__global__ void k_ehid(const float* __restrict__ efeat, const float* __restrict__ w,
                       const float* __restrict__ b, _Float16* __restrict__ hmid16, int E) {
    long i = (long)blockIdx.x * blockDim.x + threadIdx.x;
    long tot = (long)E * KH;
    for (; i < tot; i += (long)gridDim.x * blockDim.x) {
        int e = (int)(i >> 7), k = (int)(i & 127);
        float s = b[k];
#pragma unroll
        for (int j = 0; j < 5; ++j) s += efeat[e * 5 + j] * w[j * KH + k];
        hmid16[i] = (_Float16)fmaxf(s, 0.0f);
    }
}

// Pack e2_w / e2_b into WMMA-B-ready f16 slabs, grouped per (khalf, otile-pair)
// so each workgroup's staging is one contiguous 266240B burst:
//   B16[kh][pair][otl][cl][n][kk]
//   kh=0: global chunk cg=cl (cl<128; cl 128/129 are zero padding)
//   kh=1: cg=128+cl (cg 256/257 are the e2_b bias chunks)
//   cg<256: k=cg>>1, i=(cg&1)*32+kk, val=e2_w[k*4096 + i*64 + o]
//   cg>=256: i=(cg-256)*32+kk,       val=e2_b[i*64 + o],  o = pair*32+otl*16+n
__global__ void k_prepB(const float* __restrict__ e2w, const float* __restrict__ e2b,
                        _Float16* __restrict__ B16) {
    long i = (long)blockIdx.x * blockDim.x + threadIdx.x;
    const long tot = 2L * 2 * 2 * NCL * CHKH;
    for (; i < tot; i += (long)gridDim.x * blockDim.x) {
        int kk  = (int)(i & 31);
        int n   = (int)((i >> 5) & 15);
        int cl  = (int)((i >> 9) % NCL);
        int rest = (int)(i / ((long)NCL * CHKH));
        int otl  = rest & 1;
        int pair = (rest >> 1) & 1;
        int kh   = rest >> 2;
        int o = pair * 32 + otl * 16 + n;
        float v = 0.0f;
        int cg = kh * 128 + cl;
        if (kh == 1 || cl < 128) {
            if (cg < 256) {
                int k = cg >> 1, ii = (cg & 1) * 32 + kk;
                v = e2w[(long)k * 4096 + ii * 64 + o];
            } else {
                int ii = (cg - 256) * 32 + kk;
                v = e2b[ii * 64 + o];
            }
        }
        B16[i] = (_Float16)v;
    }
}

__global__ void k_deg_acc(const int* __restrict__ dst, float* __restrict__ deg, int E) {
    long i = (long)blockIdx.x * blockDim.x + threadIdx.x;
    for (; i < E; i += (long)gridDim.x * blockDim.x) atomicAdd(&deg[dst[i]], 1.0f);
}
__global__ void k_deg_inv(const float* __restrict__ deg, float* __restrict__ dinv, int N) {
    long i = (long)blockIdx.x * blockDim.x + threadIdx.x;
    for (; i < N; i += (long)gridDim.x * blockDim.x) dinv[i] = 1.0f / fmaxf(deg[i], 1.0f);
}

// ---------------------------------------------------------------------------
// Hot kernel: fused edge message GEMM (WMMA f16) + scatter-add aggregation.
// grid = (64, 2, 2): y = output 16-col tile PAIR, z = K half.
// LDS: 2 otiles x 130 chunks x 1KB = 260KB slab, staged via async-to-LDS.
// ---------------------------------------------------------------------------
__global__ void __launch_bounds__(512)
k_msg_wmma(const float* __restrict__ h, const _Float16* __restrict__ hmid16,
           const _Float16* __restrict__ B16, const int* __restrict__ src,
           const int* __restrict__ dst, float* __restrict__ agg, int nblk) {
    extern __shared__ _Float16 sB[];                 // 2*NCL*CHKH f16 = 266240B

    const int pair = blockIdx.y;                     // output cols [pair*32, +32)
    const int kh   = blockIdx.z;                     // K half
    const int tid  = threadIdx.x;

    // --- stage this workgroup's B slab into LDS with gfx1250 async loads ---
    {
        const int slabBytes = 2 * NCL * CHKH * 2;    // 266240
        const unsigned long long gBase =
            (unsigned long long)(const void*)(B16 + (long)(kh * 2 + pair) * 2 * NCL * CHKH);
        const unsigned ldsBase = (unsigned)(unsigned long long)(const void*)sB; // low 32b = LDS offset
        for (int byt = tid * 16; byt < slabBytes; byt += blockDim.x * 16) {
            unsigned la = ldsBase + (unsigned)byt;
            unsigned ga = (unsigned)byt;
            asm volatile("global_load_async_to_lds_b128 %0, %1, %2"
                         :: "v"(la), "v"(ga), "s"(gBase) : "memory");
        }
        asm volatile("s_wait_asynccnt 0x0" ::: "memory");
    }
    __syncthreads();

    const int wave = tid >> 5;                       // 16 waves
    const int lane = tid & 31;                       // wave32
    const int row  = lane & 15;                      // A row / B col / C col (ISA layouts)
    const int hiK  = lane >> 4;                      // K-half selector within operands
    const int base0 = hiK * 8;

    // Two stable LDS base pointers (otl=1 offsets exceed the 16-bit ds imm)
    const _Float16* sBl0 = sB + row * 32 + hiK * 8;
    const _Float16* sBl1 = sBl0 + (long)NCL * CHKH;

    for (int blk = blockIdx.x * 16 + wave; blk < nblk; blk += gridDim.x * 16) {
        const int eBase   = blk * 16;
        const int rowEdge = eBase + row;
        const int s       = src[rowEdge];
        const float* xr   = h + (long)s * HID;

        // Per-lane resident x slices in f16, matching the 16-bit A 16x32 layout:
        // xs[half][g][j] = x[row, half*32 + g*16 + base0 + j]
        v8h xs[2][2];
#pragma unroll
        for (int half = 0; half < 2; ++half)
#pragma unroll
            for (int g = 0; g < 2; ++g) {
                const float* p = xr + half * 32 + g * 16 + base0;
                v8h t;
#pragma unroll
                for (int j = 0; j < 8; ++j) t[j] = (_Float16)p[j];
                xs[half][g] = t;
            }

        const v8h* hm = (const v8h*)(hmid16 + (long)rowEdge * KH);

        v8f acc[2];
#pragma unroll
        for (int otl = 0; otl < 2; ++otl)
#pragma unroll
            for (int r = 0; r < 8; ++r) acc[otl][r] = 0.0f;

        for (int k8l = 0; k8l < 8; ++k8l) {          // 64 k-values in this K half
            v8h hv = hm[kh * 8 + k8l];               // 8 hmid scalars (16B load)
#pragma unroll
            for (int kk = 0; kk < 8; ++kk) {
                const _Float16 sc = hv[kk];
                const long cb = (long)((k8l * 8 + kk) * 2) * CHKH;
                // Build BOTH half-A operands up front...
                v8h a00 = scale8(xs[0][0], sc);
                v8h a01 = scale8(xs[0][1], sc);
                v8h a10 = scale8(xs[1][0], sc);
                v8h a11 = scale8(xs[1][1], sc);
                v16h A0 = __builtin_shufflevector(a00, a01,
                    0,1,2,3,4,5,6,7,8,9,10,11,12,13,14,15);
                v16h A1 = __builtin_shufflevector(a10, a11,
                    0,1,2,3,4,5,6,7,8,9,10,11,12,13,14,15);
                // ...and pin them there: VALU may not sink / WMMA may not hoist
                // across this point (DS-read, VMEM-read, SALU stay free to
                // cross for prefetch scheduling). Forces A0/A1 into distinct
                // registers; next iteration's pk_muls fill the WAR window.
                __builtin_amdgcn_sched_barrier(SB_MASK);
                acc[0] = __builtin_amdgcn_wmma_f32_16x16x32_f16(
                    false, A0, false, ldB16(sBl0 + cb), (short)0, acc[0], false, false);
                acc[1] = __builtin_amdgcn_wmma_f32_16x16x32_f16(
                    false, A0, false, ldB16(sBl1 + cb), (short)0, acc[1], false, false);
                acc[0] = __builtin_amdgcn_wmma_f32_16x16x32_f16(
                    false, A1, false, ldB16(sBl0 + cb + CHKH), (short)0, acc[0], false, false);
                acc[1] = __builtin_amdgcn_wmma_f32_16x16x32_f16(
                    false, A1, false, ldB16(sBl1 + cb + CHKH), (short)0, acc[1], false, false);
            }
        }
        // Bias chunks (e2_b folded in as local chunks 128/129, unit scale) - K half 1 only
        if (kh == 1) {
            v16h A0 = __builtin_shufflevector(xs[0][0], xs[0][1],
                0,1,2,3,4,5,6,7,8,9,10,11,12,13,14,15);
            v16h A1 = __builtin_shufflevector(xs[1][0], xs[1][1],
                0,1,2,3,4,5,6,7,8,9,10,11,12,13,14,15);
            const long cb = (long)128 * CHKH;
            __builtin_amdgcn_sched_barrier(SB_MASK);
            acc[0] = __builtin_amdgcn_wmma_f32_16x16x32_f16(
                false, A0, false, ldB16(sBl0 + cb), (short)0, acc[0], false, false);
            acc[1] = __builtin_amdgcn_wmma_f32_16x16x32_f16(
                false, A0, false, ldB16(sBl1 + cb), (short)0, acc[1], false, false);
            acc[0] = __builtin_amdgcn_wmma_f32_16x16x32_f16(
                false, A1, false, ldB16(sBl0 + cb + CHKH), (short)0, acc[0], false, false);
            acc[1] = __builtin_amdgcn_wmma_f32_16x16x32_f16(
                false, A1, false, ldB16(sBl1 + cb + CHKH), (short)0, acc[1], false, false);
        }

        // Fused scatter-aggregate. C layout: VGPR r -> edge row r + hiK*8, col = lane&15.
        // K-half partial sums merge here (additive).
#pragma unroll
        for (int r = 0; r < 8; ++r) {
            const int eRow = eBase + r + hiK * 8;
            const int d    = dst[eRow];
#pragma unroll
            for (int otl = 0; otl < 2; ++otl)
                atomicAdd(&agg[(long)d * HID + pair * 32 + otl * 16 + row], acc[otl][r]);
        }
    }
}

// ---------------------------------------------------------------------------
// GRU cell, weights LDS-resident (96KB dynamic shared), one node per pass.
// ---------------------------------------------------------------------------
__global__ void __launch_bounds__(256)
k_gru(const float* __restrict__ agg, const float* __restrict__ dinv,
      const float* __restrict__ cbias, const float* __restrict__ w_ih,
      const float* __restrict__ w_hh, const float* __restrict__ b_ih,
      const float* __restrict__ b_hh, float* __restrict__ h, int N) {
    extern __shared__ float sW[];                    // [64*192] ih | [64*192] hh
    float* sWih = sW;
    float* sWhh = sW + 64 * 192;
    __shared__ float sm[64], shv[64], sgi[192], sgh[192];

    for (int i = threadIdx.x; i < 64 * 192; i += blockDim.x) {
        sWih[i] = w_ih[i];
        sWhh[i] = w_hh[i];
    }
    __syncthreads();

    const int t = threadIdx.x;
    for (int node = blockIdx.x; node < N; node += gridDim.x) {
        if (t < 64) {
            sm[t]  = fmaxf(agg[(long)node * HID + t] * dinv[node] + cbias[t], 0.0f);
            shv[t] = h[(long)node * HID + t];
        }
        __syncthreads();
        if (t < 192) {
            float s1 = b_ih[t], s2 = b_hh[t];
#pragma unroll 8
            for (int i = 0; i < 64; ++i) {
                s1 += sm[i]  * sWih[i * 192 + t];
                s2 += shv[i] * sWhh[i * 192 + t];
            }
            sgi[t] = s1;
            sgh[t] = s2;
        }
        __syncthreads();
        if (t < 64) {
            float r = sigm(sgi[t] + sgh[t]);
            float z = sigm(sgi[64 + t] + sgh[64 + t]);
            float n = tanhf(sgi[128 + t] + r * sgh[128 + t]);
            h[(long)node * HID + t] = (1.0f - z) * n + z * shv[t];
        }
        __syncthreads();
    }
}

// ---------------------------------------------------------------------------
// Set2Set pooling
// ---------------------------------------------------------------------------
__device__ __forceinline__ unsigned orderKey(float x) {
    unsigned u = __float_as_uint(x);
    return (u & 0x80000000u) ? ~u : (u | 0x80000000u);
}
__device__ __forceinline__ float keyToFloat(unsigned k) {
    unsigned u = (k & 0x80000000u) ? (k ^ 0x80000000u) : ~k;
    return __uint_as_float(u);
}

__global__ void k_s2s_init(float* q_star, float* hl, float* cl) {
    int t = threadIdx.x;
    if (t < 128) q_star[t] = 0.0f;
    if (t < 64) { hl[t] = 0.0f; cl[t] = 0.0f; }
}

__global__ void k_lstm(const float* __restrict__ q_star, const float* __restrict__ w_ih,
                       const float* __restrict__ w_hh, const float* __restrict__ b_ih,
                       const float* __restrict__ b_hh, float* hl, float* cl, float* qv,
                       float* readout, float* sumExp, unsigned* maxkey) {
    __shared__ float g[256];
    const int t = threadIdx.x;
    float s = b_ih[t] + b_hh[t];
    for (int i = 0; i < 128; ++i) s += q_star[i] * w_ih[i * 256 + t];
    for (int i = 0; i < 64; ++i)  s += hl[i]     * w_hh[i * 256 + t];
    g[t] = s;
    __syncthreads();
    if (t < 64) {
        float ig = sigm(g[t]), fg = sigm(g[64 + t]);
        float gg = tanhf(g[128 + t]), og = sigm(g[192 + t]);
        float c = fg * cl[t] + ig * gg;
        cl[t] = c;
        float hh = og * tanhf(c);
        hl[t] = hh;
        qv[t] = hh;
        readout[t] = 0.0f;
    }
    if (t == 0) { *maxkey = 0u; *sumExp = 0.0f; }
}

__global__ void k_dote(const float* __restrict__ h, const float* __restrict__ qv,
                       float* __restrict__ evals, unsigned* maxkey, int N) {
    long nd = (long)blockIdx.x * blockDim.x + threadIdx.x;
    for (; nd < N; nd += (long)gridDim.x * blockDim.x) {
        const float* hr = h + nd * HID;
        float s = 0.0f;
#pragma unroll 8
        for (int i = 0; i < HID; ++i) s += hr[i] * qv[i];
        evals[nd] = s;
        atomicMax(maxkey, orderKey(s));
    }
}

__global__ void k_expsum(float* __restrict__ evals, const unsigned* maxkey,
                         float* sumExp, int N) {
    const float mv = keyToFloat(*maxkey);
    float loc = 0.0f;
    long nd = (long)blockIdx.x * blockDim.x + threadIdx.x;
    for (; nd < N; nd += (long)gridDim.x * blockDim.x) {
        float w = __expf(evals[nd] - mv);
        evals[nd] = w;
        loc += w;
    }
    atomicAdd(sumExp, loc);
}

__global__ void k_readout(const float* __restrict__ h, const float* __restrict__ evals,
                          float* __restrict__ readout, int N) {
    __shared__ float red[256];
    const int col = threadIdx.x & 63;
    const int rr  = threadIdx.x >> 6;                 // 0..3
    float acc = 0.0f;
    for (int nd = blockIdx.x * 4 + rr; nd < N; nd += gridDim.x * 4)
        acc += evals[nd] * h[(long)nd * HID + col];
    red[threadIdx.x] = acc;
    __syncthreads();
    if (rr == 0) {
        float s = red[col] + red[64 + col] + red[128 + col] + red[192 + col];
        atomicAdd(&readout[col], s);
    }
}

__global__ void k_s2s_fin(float* q_star, const float* qv, const float* readout,
                          const float* sumExp) {
    const int t = threadIdx.x;
    if (t < 64)       q_star[t] = qv[t];
    else if (t < 128) q_star[t] = readout[t - 64] / *sumExp;
}

__global__ void k_copy_out(const float* __restrict__ q_star, const float* __restrict__ h,
                           float* __restrict__ out, long tot) {
    long i = (long)blockIdx.x * blockDim.x + threadIdx.x;
    for (; i < tot; i += (long)gridDim.x * blockDim.x)
        out[i] = (i < 128) ? q_star[i] : h[i - 128];
}

// ---------------------------------------------------------------------------
// Host launcher
// ---------------------------------------------------------------------------
extern "C" void kernel_launch(void* const* d_in, const int* in_sizes, int n_in,
                              void* d_out, int out_size, void* d_ws, size_t ws_size,
                              hipStream_t stream) {
    const float* nfeat  = (const float*)d_in[0];
    const float* efeat  = (const float*)d_in[1];
    const int*   src    = (const int*)d_in[2];
    const int*   dst    = (const int*)d_in[3];
    const float* lin0_w = (const float*)d_in[4];
    const float* lin0_b = (const float*)d_in[5];
    const float* e1_w   = (const float*)d_in[6];
    const float* e1_b   = (const float*)d_in[7];
    const float* e2_w   = (const float*)d_in[8];
    const float* e2_b   = (const float*)d_in[9];
    const float* cbias  = (const float*)d_in[10];
    const float* gW_ih  = (const float*)d_in[11];
    const float* gW_hh  = (const float*)d_in[12];
    const float* gb_ih  = (const float*)d_in[13];
    const float* gb_hh  = (const float*)d_in[14];
    const float* lW_ih  = (const float*)d_in[15];
    const float* lW_hh  = (const float*)d_in[16];
    const float* lb_ih  = (const float*)d_in[17];
    const float* lb_hh  = (const float*)d_in[18];
    (void)n_in; (void)ws_size; (void)out_size;

    const int N = in_sizes[0] / 16;
    const int E = in_sizes[1] / 5;

    // Workspace carve-up (~62 MB total)
    char* wp = (char*)d_ws;
    auto carve = [&](size_t bytes) -> char* {
        char* p = wp;
        wp += (bytes + 255) & ~(size_t)255;
        return p;
    };
    float*     h      = (float*)carve((size_t)N * HID * 4);
    float*     agg    = (float*)carve((size_t)N * HID * 4);
    float*     deg    = (float*)carve((size_t)N * 4);
    float*     dinv   = (float*)carve((size_t)N * 4);
    _Float16*  hmid16 = (_Float16*)carve((size_t)E * KH * 2);
    _Float16*  B16    = (_Float16*)carve((size_t)8 * NCL * CHKH * 2);
    float*     evals  = (float*)carve((size_t)N * 4);
    float*     scal   = (float*)carve(512 * 4);
    float*     q_star  = scal;
    float*     hl      = scal + 128;
    float*     cl      = scal + 192;
    float*     qv      = scal + 256;
    float*     ro      = scal + 320;
    float*     sumExp  = scal + 384;
    unsigned*  maxkey  = (unsigned*)(scal + 385);

    // --- one-time prep ---
    k_lin0<<<512, 256, 0, stream>>>(nfeat, lin0_w, lin0_b, h, N);
    k_ehid<<<1024, 256, 0, stream>>>(efeat, e1_w, e1_b, hmid16, E);
    k_prepB<<<(8 * NCL * CHKH + 255) / 256, 256, 0, stream>>>(e2_w, e2_b, B16);
    k_zero_f32<<<256, 256, 0, stream>>>(deg, N);
    k_deg_acc<<<512, 256, 0, stream>>>(dst, deg, E);
    k_deg_inv<<<256, 256, 0, stream>>>(deg, dinv, N);

    // --- 3 message-passing rounds ---
    const int nblk = E / 16;                          // E=100000 -> 6250 full blocks
    const size_t smemB = (size_t)2 * NCL * CHKH * 2;  // 260KB B-slab in LDS
    const size_t smemG = (size_t)2 * 64 * 192 * 4;    // 96KB GRU weights in LDS
    for (int it = 0; it < 3; ++it) {
        k_zero_f32<<<512, 256, 0, stream>>>(agg, (long)N * HID);
        k_msg_wmma<<<dim3(64, 2, 2), 512, smemB, stream>>>(h, hmid16, B16, src, dst, agg, nblk);
        k_gru<<<256, 256, smemG, stream>>>(agg, dinv, cbias, gW_ih, gW_hh, gb_ih, gb_hh, h, N);
    }

    // --- Set2Set (3 rounds) ---
    k_s2s_init<<<1, 256, 0, stream>>>(q_star, hl, cl);
    for (int it = 0; it < 3; ++it) {
        k_lstm<<<1, 256, 0, stream>>>(q_star, lW_ih, lW_hh, lb_ih, lb_hh,
                                      hl, cl, qv, ro, sumExp, maxkey);
        k_dote<<<80, 256, 0, stream>>>(h, qv, evals, maxkey, N);
        k_expsum<<<80, 256, 0, stream>>>(evals, maxkey, sumExp, N);
        k_readout<<<80, 256, 0, stream>>>(h, evals, ro, N);
        k_s2s_fin<<<1, 128, 0, stream>>>(q_star, qv, ro, sumExp);
    }

    // --- output: [q_star (128) | out (N*64)] ---
    k_copy_out<<<1024, 256, 0, stream>>>(q_star, h, (float*)d_out, 128 + (long)N * HID);
}
